// MultiParticleVortexNetwork_54915451846787
// MI455X (gfx1250) — compile-verified, matrix operand order checked by reading, not run on hardware
//
#include <hip/hip_runtime.h>
#include <hip/hip_bf16.h>

// ---------------------------------------------------------------------------
// MultiParticleVortexNetwork on gfx1250 (MI455X, wave32, WMMA)
//   Phase 1: O(B*N^2) pairwise vortex velocity + analytic Jacobian
//            (VALU/transcendental bound; j-loop split 4-way + shfl reduce)
//   Phase 2: MLP 10->100->6 via v_wmma_f32_16x16x32_f16, weights pre-swizzled
//            into per-lane fragment order (b128 fragment loads)
// ---------------------------------------------------------------------------

#define B_   16
#define N_   512
#define HID_ 100
#define ROWS (B_ * N_)          // 8192
#define KP1  32                 // padded K for layer 1 (10 -> 32)
#define NT1  7                  // N tiles for layer 1 (100 -> 7*16 = 112)
#define KP2  128                // padded K for layer 2 (100 -> 128)

#define EPS_        1e-6f
#define INV_TWO_PI  0.15915494309189535f
#define NEG_SLOPE   0.1f

typedef __attribute__((ext_vector_type(16))) _Float16 v16h;
typedef __attribute__((ext_vector_type(8)))  _Float16 v8h;
typedef __attribute__((ext_vector_type(8)))  float    v8f;

union U16h {
    v16h     v;
    v8h      v8[2];
    _Float16 h[16];
};

// Fragment K index for slot t in a lane's 16-half A/B fragment
// (16-bit 16x32 layout: lanes 0-15 hold K 0-7 & 16-23; lanes 16-31 K 8-15 & 24-31)
__device__ __forceinline__ int frag_k(int hp, int t) {
    return hp * 8 + ((t < 8) ? t : (8 + t));   // t<8 -> hp*8+t ; t>=8 -> hp*8+16+(t-8)
}

// ---------------------------------------------------------------------------
// Kernel 0: pack weights directly into per-lane WMMA fragment order.
//   W1f: [nt][lane][16]  (7*32*16 halves)   B-fragment for layer 1
//   W2f: [kk][lane][16]  (4*32*16 halves)   B-fragment for layer 2
// ---------------------------------------------------------------------------
__global__ void pack_weights_kernel(const float* __restrict__ W1,
                                    const float* __restrict__ W2,
                                    _Float16* __restrict__ W1f,
                                    _Float16* __restrict__ W2f) {
    int idx = blockIdx.x * blockDim.x + threadIdx.x;
    const int n1 = NT1 * 32 * 16;           // 3584
    const int n2 = 4 * 32 * 16;             // 2048
    if (idx < n1) {
        int nt   = idx >> 9;                // /512
        int rem  = idx & 511;
        int lane = rem >> 4;
        int t    = rem & 15;
        int hp = lane >> 4, lr = lane & 15;
        int k = frag_k(hp, t);              // 0..31
        int n = nt * 16 + lr;               // 0..111
        float v = (k < 10 && n < HID_) ? W1[k * HID_ + n] : 0.0f;
        W1f[idx] = (_Float16)v;
    } else if (idx < n1 + n2) {
        int j    = idx - n1;
        int kk   = j >> 9;
        int rem  = j & 511;
        int lane = rem >> 4;
        int t    = rem & 15;
        int hp = lane >> 4, lr = lane & 15;
        int k = kk * 32 + frag_k(hp, t);    // 0..127
        int n = lr;                         // 0..15
        float v = (k < HID_ && n < 6) ? W2[k * 6 + n] : 0.0f;
        W2f[j] = (_Float16)v;
    }
}

// ---------------------------------------------------------------------------
// Kernel 1: zero the H activation buffer (padding columns must be exactly 0)
// ---------------------------------------------------------------------------
__global__ void zero_h_kernel(uint4* __restrict__ H) {
    int idx = blockIdx.x * blockDim.x + threadIdx.x;   // 131072 uint4
    H[idx] = make_uint4(0u, 0u, 0u, 0u);
}

// ---------------------------------------------------------------------------
// Kernel 2: pairwise induced velocity + analytic Jacobian -> f16 feature rows
//   grid = B * (N/64) = 128 blocks of 256 threads
//   4 threads per particle, each covering 128 of 512 j's; shfl_xor reduce
// ---------------------------------------------------------------------------
__global__ __launch_bounds__(256) void pairwise_kernel(
        const float* __restrict__ inp,     // (B,N,6)
        _Float16* __restrict__ X) {        // (ROWS, KP1) f16
    __shared__ float s_y[N_], s_x[N_], s_tau[N_], s_is2[N_], s_c[N_], s_d[N_];

    const int tid  = threadIdx.x;
    const int b    = blockIdx.x >> 3;          // / 8
    const int part = blockIdx.x & 7;           // i-tile of 64
    const int i    = part * 64 + (tid >> 2);   // particle index
    const int jsub = tid & 3;                  // j quarter

    const float* base = inp + (size_t)b * N_ * 6;

    // cooperative LDS fill: each thread loads 2 particles
    #pragma unroll
    for (int rep = 0; rep < 2; ++rep) {
        int j = tid + rep * 256;
        const float* p = base + j * 6;
        s_y[j]   = p[0];
        s_x[j]   = p[1];
        s_tau[j] = p[2];
        float sg = p[3];
        s_is2[j] = 1.0f / (sg * sg);
        s_c[j]   = p[4];
        s_d[j]   = p[5];
    }
    __syncthreads();

    const float* pi = base + i * 6;
    const float yi = pi[0], xi = pi[1];
    const float tau_i = pi[2], sig_i = pi[3], c_i = pi[4], d_i = pi[5];

    float vy = 0.f, vx = 0.f;
    float J00 = 0.f, J01 = 0.f, J10 = 0.f, J11 = 0.f;

    const int j0 = jsub * 128;
    #pragma unroll 4
    for (int jj = 0; jj < 128; ++jj) {
        const int j = j0 + jj;
        float dy = yi - s_y[j];
        float dx = xi - s_x[j];
        float sq = dy * dy + dx * dx;
        float se = sq + EPS_;
        float inv_se = 1.0f / se;
        float rinv = __frsqrt_rn(se);
        float r = se * rinv;

        float is2 = s_is2[j];
        float eA  = __expf(-sq * is2);         // exp(-sq/sig^2)
        float A   = 1.0f - eA;
        float Bx  = __expf(-s_c[j] * r - s_d[j] * sq);

        float pref = s_tau[j] * INV_TWO_PI * inv_se;
        float g    = pref * A * Bx;            // falloff

        // d(falloff)/d(sq)
        float Ap = eA * is2;                                   // dA/du
        float Bp = Bx * (-(0.5f * s_c[j]) * rinv - s_d[j]);    // dB/du
        float dgdu = pref * (Ap * Bx + A * Bp) - g * inv_se;

        float msk = (j != i) ? 1.0f : 0.0f;
        g    *= msk;
        dgdu *= msk;

        vy  += -g * dx;
        vx  +=  g * dy;

        float t2 = 2.0f * dgdu;
        J00 += -dx * t2 * dy;          // d vy / d y
        J01 += -g - dx * dx * t2;      // d vy / d x
        J10 +=  g + dy * dy * t2;      // d vx / d y
        J11 +=  dy * dx * t2;          // d vx / d x
    }

    // butterfly reduce across the 4 lanes that share this particle (same wave)
    #pragma unroll
    for (int m = 1; m < 4; m <<= 1) {
        vy  += __shfl_xor(vy,  m);
        vx  += __shfl_xor(vx,  m);
        J00 += __shfl_xor(J00, m);
        J01 += __shfl_xor(J01, m);
        J10 += __shfl_xor(J10, m);
        J11 += __shfl_xor(J11, m);
    }

    if (jsub == 0) {
        // feature row: [tau, sig, c, d, vy, vx, J00, J01, J10, J11, 0...0]
        U16h fr;
        fr.h[0] = (_Float16)tau_i;  fr.h[1] = (_Float16)sig_i;
        fr.h[2] = (_Float16)c_i;    fr.h[3] = (_Float16)d_i;
        fr.h[4] = (_Float16)vy;     fr.h[5] = (_Float16)vx;
        fr.h[6] = (_Float16)J00;    fr.h[7] = (_Float16)J01;
        fr.h[8] = (_Float16)J10;    fr.h[9] = (_Float16)J11;
        #pragma unroll
        for (int t = 10; t < 16; ++t) fr.h[t] = (_Float16)0.0f;

        const int row = b * N_ + i;
        _Float16* xr = X + (size_t)row * KP1;
        *(v16h*)xr = fr.v;                         // halves 0..15
        *(v16h*)(xr + 16) = (v16h)(_Float16)0.0f;  // halves 16..31 zero
    }
}

// ---------------------------------------------------------------------------
// Kernel 3: layer 1 GEMM  H = LeakyReLU(X @ W1 + b1)
//   one wave per 16x16 output tile; single v_wmma_f32_16x16x32_f16 (K=32)
//   grid = (ROWS/16, NT1) = (512, 7), block = 32
// ---------------------------------------------------------------------------
__global__ __launch_bounds__(32) void gemm1_kernel(
        const _Float16* __restrict__ X,     // (ROWS, KP1)
        const _Float16* __restrict__ W1f,   // per-lane fragments [nt][lane][16]
        const float* __restrict__ b1,       // (HID_)
        _Float16* __restrict__ H) {         // (ROWS, KP2)
    const int lane = threadIdx.x;
    const int m0 = blockIdx.x * 16;
    const int nt = blockIdx.y;
    const int hp = lane >> 4;        // 0|1 -> K group
    const int lr = lane & 15;

    // A fragment: row m0+lr, halves [hp*8 .. +7] and [hp*8+16 .. +7]
    U16h a;
    const _Float16* arow = X + (size_t)(m0 + lr) * KP1 + hp * 8;
    a.v8[0] = *(const v8h*)(arow);
    a.v8[1] = *(const v8h*)(arow + 16);

    // B fragment: pre-swizzled, contiguous per lane
    U16h bf;
    bf.v = *(const v16h*)(W1f + ((size_t)nt * 32 + lane) * 16);

    v8f c = {};
    c = __builtin_amdgcn_wmma_f32_16x16x32_f16(false, a.v, false, bf.v,
                                               (short)0, c, false, false);

    const int nn = nt * 16 + lr;
    const float bias = (nn < HID_) ? b1[nn] : 0.0f;
    #pragma unroll
    for (int r = 0; r < 8; ++r) {
        int m = m0 + r + hp * 8;               // D layout: lanes 16-31 -> M+8
        float h = c[r] + bias;
        h = (h >= 0.0f) ? h : NEG_SLOPE * h;   // LeakyReLU(0.1)
        H[(size_t)m * KP2 + nn] = (_Float16)h; // cols >= 112 stay 0 (pre-zeroed)
    }
}

// ---------------------------------------------------------------------------
// Kernel 4: layer 2 GEMM + output update
//   out = stack(y+.1dy, x+.1dx, tau+.1dtau, sig+.1dsig, .1*sp(dc), .1*sp(dd))
//   one wave per 16-row block; 4 accumulating WMMAs over K=128
// ---------------------------------------------------------------------------
__global__ __launch_bounds__(32) void gemm2_kernel(
        const _Float16* __restrict__ H,     // (ROWS, KP2)
        const _Float16* __restrict__ W2f,   // per-lane fragments [kk][lane][16]
        const float* __restrict__ b2,       // (6)
        const float* __restrict__ inp,      // (B,N,6)
        float* __restrict__ out) {          // (B,N,6)
    const int lane = threadIdx.x;
    const int m0 = blockIdx.x * 16;
    const int hp = lane >> 4;
    const int lr = lane & 15;

    v8f c = {};
    #pragma unroll
    for (int kk = 0; kk < 4; ++kk) {
        U16h a, bf;
        const _Float16* arow = H + (size_t)(m0 + lr) * KP2 + kk * 32 + hp * 8;
        a.v8[0] = *(const v8h*)(arow);
        a.v8[1] = *(const v8h*)(arow + 16);
        bf.v = *(const v16h*)(W2f + ((size_t)kk * 32 + lane) * 16);
        c = __builtin_amdgcn_wmma_f32_16x16x32_f16(false, a.v, false, bf.v,
                                                   (short)0, c, false, false);
    }

    if (lr < 6) {
        const float bias = b2[lr];
        #pragma unroll
        for (int r = 0; r < 8; ++r) {
            int row = m0 + r + hp * 8;
            float o = c[r] + bias;
            float res;
            if (lr < 4) {
                res = inp[(size_t)row * 6 + lr] + 0.1f * o;     // y/x/tau/sig
            } else {
                float sp = (o > 20.0f) ? o : log1pf(__expf(o)); // softplus
                res = 0.1f * sp;                                // c/d update
            }
            out[(size_t)row * 6 + lr] = res;
        }
    }
}

// ---------------------------------------------------------------------------
// launcher
// ---------------------------------------------------------------------------
extern "C" void kernel_launch(void* const* d_in, const int* in_sizes, int n_in,
                              void* d_out, int out_size, void* d_ws, size_t ws_size,
                              hipStream_t stream) {
    (void)in_sizes; (void)n_in; (void)out_size; (void)ws_size;

    const float* inp = (const float*)d_in[0];   // (B,N,6)
    const float* W1  = (const float*)d_in[1];   // (10,100)
    const float* b1  = (const float*)d_in[2];   // (100)
    const float* W2  = (const float*)d_in[3];   // (100,6)
    const float* b2  = (const float*)d_in[4];   // (6)
    float* out = (float*)d_out;                 // (B,N,6)

    char* ws = (char*)d_ws;
    _Float16* X   = (_Float16*)(ws);                                  // 512 KB
    _Float16* H   = (_Float16*)(ws + (size_t)ROWS * KP1 * 2);         // 2 MB
    _Float16* W1f = (_Float16*)(ws + (size_t)ROWS * KP1 * 2
                                   + (size_t)ROWS * KP2 * 2);         // 7 KB
    _Float16* W2f = W1f + NT1 * 32 * 16;                              // 4 KB

    pack_weights_kernel<<<(NT1 * 32 * 16 + 4 * 32 * 16 + 255) / 256, 256, 0,
                          stream>>>(W1, W2, W1f, W2f);

    zero_h_kernel<<<(ROWS * KP2 * 2 / 16) / 256, 256, 0, stream>>>((uint4*)H);

    pairwise_kernel<<<B_ * (N_ / 64), 256, 0, stream>>>(inp, X);

    gemm1_kernel<<<dim3(ROWS / 16, NT1), 32, 0, stream>>>(X, W1f, b1, H);

    gemm2_kernel<<<ROWS / 16, 32, 0, stream>>>(H, W2f, b2, inp, out);
}
